// SelectiveDLinOSSLayer_76682346102971
// MI455X (gfx1250) — matrix operand order repaired and checked by reading, not compiled, not run on hardware
//
#include <hip/hip_runtime.h>

// ---------------- problem constants ----------------
#define HD    768      // hidden dim
#define PD    384      // ssm size
#define LENT  2048     // sequence length
#define BSZ   16       // batch
#define NTOK  (BSZ*LENT)        // 32768 tokens
#define PI_F  3.14159265358979f

// ---------------- CDNA5 WMMA types ----------------
typedef __attribute__((ext_vector_type(16))) __bf16 v16bf;
typedef __attribute__((ext_vector_type(8)))  float  v8f;
typedef __attribute__((ext_vector_type(4)))  unsigned int v4u;
typedef __attribute__((ext_vector_type(8)))  int v8i;
typedef __attribute__((ext_vector_type(4)))  int v4i;

union ABf { uint4 q[2]; v16bf v; };

static __device__ inline v8f wmma_bf16(const ABf& a, const ABf& b, v8f c) {
  return __builtin_amdgcn_wmma_f32_16x16x32_bf16(
      /*neg_a=*/false, a.v, /*neg_b=*/false, b.v,
      /*c_mod=*/(short)0, c, /*reuse_a=*/false, /*reuse_b=*/false);
}

// ---------------- Tensor Data Mover (guarded) ----------------
// 2D tile load, D# per CDNA5 ISA:
//  group0: count=1, lds_addr[63:32], global_addr[120:64], type=2[127:126]
//  group1: data_size=2B, tensor_dim0/1, tile_dim0/1, tensor_dim0_stride
// Toolchain probe: 6-arg builtin (g0, g1, g2, g3, g4_extra, cpol); tensors
// here are 2D so groups 2/3 and the extra group are all-zero.
#if __has_builtin(__builtin_amdgcn_tensor_load_to_lds)
#define USE_TDM 1
#else
#define USE_TDM 0
#endif

#if USE_TDM
static __device__ inline void tdm_load_2d(unsigned lds_off, const void* gptr,
                                          unsigned tile_w, unsigned tile_h,
                                          unsigned stride_elems) {
  unsigned long long ga = (unsigned long long)(uintptr_t)gptr;
  const unsigned tens_w = stride_elems;     // row length (elements)
  const unsigned tens_h = 1u << 20;         // plenty of rows; all accesses in-bounds
  v4u g0;
  g0.x = 1u;                                              // count=1, user mode
  g0.y = lds_off;                                         // lds_addr (bytes)
  g0.z = (unsigned)(ga & 0xFFFFFFFFu);                    // global_addr[31:0]
  g0.w = (unsigned)((ga >> 32) & 0x01FFFFFFu) | (2u << 30); // addr[56:32] | type=2
  v8i g1;
  g1[0] = (int)(1u << 16);                                // data_size=1 (2 bytes)
  g1[1] = (int)((tens_w & 0xFFFFu) << 16);                // tensor_dim0[15:0]
  g1[2] = (int)((tens_w >> 16) | ((tens_h & 0xFFFFu) << 16)); // dim0 hi | dim1 lo
  g1[3] = (int)((tens_h >> 16) | (tile_w << 16));         // dim1 hi | tile_dim0
  g1[4] = (int)tile_h;                                    // tile_dim1 (tile_dim2=0)
  g1[5] = (int)stride_elems;                              // tensor_dim0_stride lo
  g1[6] = 0;                                              // stride hi | dim1_stride lo
  g1[7] = 0;
  v4i z4 = {0, 0, 0, 0};
  v8i z8 = {0, 0, 0, 0, 0, 0, 0, 0};
  __builtin_amdgcn_tensor_load_to_lds(g0, g1, z4, z4, z8, 0);
}
#endif

// ---------------- small helpers ----------------
static __device__ inline unsigned short f2bf(float f) {
  union { float f; unsigned u; } c; c.f = f;
  unsigned u = c.u;
  unsigned r = u + 0x7FFFu + ((u >> 16) & 1u);   // round-to-nearest-even
  return (unsigned short)(r >> 16);
}
static __device__ inline float bf2f(unsigned short h) {
  union { unsigned u; float f; } c; c.u = ((unsigned)h) << 16;
  return c.f;
}
static __device__ inline float sigmoidf(float x) { return 1.0f / (1.0f + __expf(-x)); }
static __device__ inline float siluf(float x)    { return x / (1.0f + __expf(-x)); }

// weight-pack region layout (elements, bf16)
#define SZWL (HD*HD)      // 589824
#define SZW  (PD*HD)      // 294912
#define O_WR (SZWL)
#define O_WT (SZWL + SZW)
#define O_WJ (SZWL + 2*SZW)
#define O_BR (SZWL + 3*SZW)
#define O_BI (SZWL + 4*SZW)
#define O_CC (SZWL + 5*SZW)
#define W_TOTAL (2*SZWL + 5*SZW)   // 2654208

// ============================================================
// Prep 1: X (f32) -> bf16, vectorized 8/thread
// ============================================================
__global__ __launch_bounds__(256) void k_cvt_x(
    const float* __restrict__ X, unsigned short* __restrict__ Xb)
{
  size_t i8 = ((size_t)blockIdx.x * 256 + threadIdx.x) * 8;
  unsigned short r[8];
  #pragma unroll
  for (int j = 0; j < 8; ++j) r[j] = f2bf(X[i8 + j]);
  *(uint4*)(Xb + i8) = *(uint4*)r;
}

// ============================================================
// Prep 2: all GEMM weights -> bf16 (Bp de-interleaved, Cc = [Cre|-Cim])
// ============================================================
__global__ __launch_bounds__(256) void k_prep_w(
    const float* __restrict__ Wl, const float* __restrict__ Wr,
    const float* __restrict__ Wth, const float* __restrict__ Wnj,
    const float* __restrict__ Bp, const float* __restrict__ Cp,
    unsigned short* __restrict__ wb)
{
  int idx = blockIdx.x * 256 + threadIdx.x;
  if (idx >= W_TOTAL) return;
  float v;
  if (idx < O_WR)       v = Wl[idx];
  else if (idx < O_WT)  v = Wr[idx - O_WR];
  else if (idx < O_WJ)  v = Wth[idx - O_WT];
  else if (idx < O_BR)  v = Wnj[idx - O_WJ];
  else if (idx < O_BI)  v = Bp[2 * (idx - O_BR)];        // B real
  else if (idx < O_CC)  v = Bp[2 * (idx - O_BI) + 1];    // B imag
  else {
    int i = idx - O_CC;
    int h = i / HD, k = i - h * HD;
    v = (k < PD) ? Cp[(size_t)h * (2 * PD) + 2 * k]
                 : -Cp[(size_t)h * (2 * PD) + 2 * (k - PD) + 1];
  }
  wb[idx] = f2bf(v);
}

// ============================================================
// Kernel 1: feats0 = silu(Xb @ Wlb^T + bl)   (N x H) bf16 out
// 64x128 tile / WG, 8 waves of 32x32; TDM double-buffered staging.
// ============================================================
__global__ __launch_bounds__(256) void k_enc_gemm(
    const unsigned short* __restrict__ Xb, const unsigned short* __restrict__ Wlb,
    const float* __restrict__ bl, unsigned short* __restrict__ feats0)
{
  __shared__ __align__(16) unsigned short sA[2][64 * 32];
  __shared__ __align__(16) unsigned short sB[2][128 * 32];
  const int m0 = blockIdx.x * 64;
  const int n0 = blockIdx.y * 128;
  const int t  = threadIdx.x;
  const int lane = t & 31, wid = t >> 5;
  const int wm = wid >> 2, wn = wid & 3;
  const int hi = lane >> 4, ln = lane & 15;

  v8f acc[2][2];
  #pragma unroll
  for (int i = 0; i < 2; ++i)
    #pragma unroll
    for (int j = 0; j < 2; ++j) acc[i][j] = (v8f){};

#if USE_TDM
  if (wid == 0) {
    tdm_load_2d((unsigned)(uintptr_t)&sA[0][0], Xb + (size_t)m0 * HD, 32, 64, HD);
    tdm_load_2d((unsigned)(uintptr_t)&sB[0][0], Wlb + (size_t)n0 * HD, 32, 128, HD);
  }
#endif
  for (int kc = 0; kc < HD / 32; ++kc) {
    const int cur = kc & 1;
    const int k0 = kc * 32;
#if USE_TDM
    if (wid == 0) {
      if (kc + 1 < HD / 32) {
        tdm_load_2d((unsigned)(uintptr_t)&sA[cur ^ 1][0],
                    Xb + (size_t)m0 * HD + k0 + 32, 32, 64, HD);
        tdm_load_2d((unsigned)(uintptr_t)&sB[cur ^ 1][0],
                    Wlb + (size_t)n0 * HD + k0 + 32, 32, 128, HD);
        __builtin_amdgcn_s_wait_tensorcnt(2);   // current buffer complete
      } else {
        __builtin_amdgcn_s_wait_tensorcnt(0);
      }
    }
    __syncthreads();
#else
    __syncthreads();
    { int m = t >> 2, c = (t & 3) * 8;
      *(uint4*)(&sA[cur][0] + m * 32 + c) =
          *(const uint4*)(Xb + (size_t)(m0 + m) * HD + k0 + c); }
    { int nn = t >> 1, c = (t & 1) * 16;
      const unsigned short* s = Wlb + (size_t)(n0 + nn) * HD + k0 + c;
      unsigned short* d = &sB[cur][0] + nn * 32 + c;
      *(uint4*)(d) = *(const uint4*)(s);
      *(uint4*)(d + 8) = *(const uint4*)(s + 8); }
    __syncthreads();
#endif
    const unsigned short* cA = &sA[cur][0];
    const unsigned short* cB = &sB[cur][0];
    ABf a[2];
    #pragma unroll
    for (int ti = 0; ti < 2; ++ti) {
      const unsigned short* pa = cA + (wm * 32 + ti * 16 + ln) * 32;
      a[ti].q[0] = *(const uint4*)(pa + hi * 8);
      a[ti].q[1] = *(const uint4*)(pa + 16 + hi * 8);
    }
    #pragma unroll
    for (int tj = 0; tj < 2; ++tj) {
      ABf b;
      const unsigned short* pb = cB + (wn * 32 + tj * 16 + ln) * 32 + hi * 8;
      b.q[0] = *(const uint4*)(pb);
      b.q[1] = *(const uint4*)(pb + 16);
      #pragma unroll
      for (int ti = 0; ti < 2; ++ti)
        acc[ti][tj] = wmma_bf16(a[ti], b, acc[ti][tj]);
    }
    __syncthreads();
  }

  #pragma unroll
  for (int tj = 0; tj < 2; ++tj) {
    int n = n0 + wn * 32 + tj * 16 + ln;
    float bias = bl[n];
    #pragma unroll
    for (int ti = 0; ti < 2; ++ti)
      #pragma unroll
      for (int v = 0; v < 8; ++v) {
        int m = m0 + wm * 32 + ti * 16 + v + hi * 8;
        feats0[(size_t)m * HD + n] = f2bf(siluf(acc[ti][tj][v] + bias));
      }
  }
}

// ============================================================
// Kernel 2: causal depthwise conv (K=4) + silu, 8 channels/thread
// ============================================================
__global__ __launch_bounds__(256) void k_conv(
    const unsigned short* __restrict__ feats0, const float* __restrict__ cw,
    const float* __restrict__ cb, unsigned short* __restrict__ feats)
{
  int idx = blockIdx.x * 256 + threadIdx.x;          // over NTOK*(HD/8)
  int hb = (idx % (HD / 8)) * 8;
  int m  = idx / (HD / 8);
  int l  = m % LENT;
  float o[8];
  #pragma unroll
  for (int i = 0; i < 8; ++i) o[i] = cb[hb + i];
  #pragma unroll
  for (int d = 0; d < 4; ++d) {                      // tap l-d uses weight j=3-d
    if (l - d >= 0) {
      uint4 q = *(const uint4*)(feats0 + (size_t)(m - d) * HD + hb);
      const unsigned short* u = (const unsigned short*)&q;
      #pragma unroll
      for (int i = 0; i < 8; ++i)
        o[i] += cw[(hb + i) * 4 + (3 - d)] * bf2f(u[i]);
    }
  }
  unsigned short r[8];
  #pragma unroll
  for (int i = 0; i < 8; ++i) r[i] = f2bf(siluf(o[i]));
  *(uint4*)(feats + (size_t)m * HD + hb) = *(uint4*)r;
}

// ============================================================
// Kernel 3: fused selective-head GEMMs (5 WMMA accumulations/tile):
//   rlin/thlin/injlin = feats @ {Wr,Wth,Winj}^T ; bu = X @ {Bre,Bim}^T
// Epilogue -> S, M=S*dt*A, SD=S*dt, gated bu (all in-register).
// ============================================================
__global__ __launch_bounds__(256) void k_sel_gemm(
    const unsigned short* __restrict__ feats, const unsigned short* __restrict__ Xb,
    const unsigned short* __restrict__ Wrb, const unsigned short* __restrict__ Wtb,
    const unsigned short* __restrict__ Wjb, const unsigned short* __restrict__ Brb,
    const unsigned short* __restrict__ Bib,
    const float* __restrict__ r_base, const float* __restrict__ th_base,
    const float* __restrict__ dt_base, const float* __restrict__ inj_logit,
    float* __restrict__ cS, float* __restrict__ cM, float* __restrict__ cSD,
    float* __restrict__ bRe, float* __restrict__ bIm)
{
  __shared__ __align__(16) unsigned short sA1[2][64 * 32];
  __shared__ __align__(16) unsigned short sA2[2][64 * 32];
  __shared__ __align__(16) unsigned short sB0[2][128 * 32];
  __shared__ __align__(16) unsigned short sB1[2][128 * 32];
  __shared__ __align__(16) unsigned short sB2[2][128 * 32];
  __shared__ __align__(16) unsigned short sB3[2][128 * 32];
  __shared__ __align__(16) unsigned short sB4[2][128 * 32];

  const int m0 = blockIdx.x * 64;
  const int p0 = blockIdx.y * 128;
  const int t  = threadIdx.x;
  const int lane = t & 31, wid = t >> 5;
  const int wm = wid >> 2, wn = wid & 3;
  const int hi = lane >> 4, ln = lane & 15;

  v8f aR[2][2], aT[2][2], aI[2][2], aBr[2][2], aBi[2][2];
  #pragma unroll
  for (int i = 0; i < 2; ++i)
    #pragma unroll
    for (int j = 0; j < 2; ++j) {
      aR[i][j] = (v8f){}; aT[i][j] = (v8f){}; aI[i][j] = (v8f){};
      aBr[i][j] = (v8f){}; aBi[i][j] = (v8f){};
    }

#if USE_TDM
  if (wid == 0) {
    tdm_load_2d((unsigned)(uintptr_t)&sA1[0][0], feats + (size_t)m0 * HD, 32, 64, HD);
    tdm_load_2d((unsigned)(uintptr_t)&sA2[0][0], Xb + (size_t)m0 * HD, 32, 64, HD);
    tdm_load_2d((unsigned)(uintptr_t)&sB0[0][0], Wrb + (size_t)p0 * HD, 32, 128, HD);
    tdm_load_2d((unsigned)(uintptr_t)&sB1[0][0], Wtb + (size_t)p0 * HD, 32, 128, HD);
    tdm_load_2d((unsigned)(uintptr_t)&sB2[0][0], Wjb + (size_t)p0 * HD, 32, 128, HD);
    tdm_load_2d((unsigned)(uintptr_t)&sB3[0][0], Brb + (size_t)p0 * HD, 32, 128, HD);
    tdm_load_2d((unsigned)(uintptr_t)&sB4[0][0], Bib + (size_t)p0 * HD, 32, 128, HD);
  }
#endif
  for (int kc = 0; kc < HD / 32; ++kc) {
    const int cur = kc & 1;
    const int k0 = kc * 32;
#if USE_TDM
    if (wid == 0) {
      if (kc + 1 < HD / 32) {
        const int kn = k0 + 32;
        tdm_load_2d((unsigned)(uintptr_t)&sA1[cur ^ 1][0], feats + (size_t)m0 * HD + kn, 32, 64, HD);
        tdm_load_2d((unsigned)(uintptr_t)&sA2[cur ^ 1][0], Xb + (size_t)m0 * HD + kn, 32, 64, HD);
        tdm_load_2d((unsigned)(uintptr_t)&sB0[cur ^ 1][0], Wrb + (size_t)p0 * HD + kn, 32, 128, HD);
        tdm_load_2d((unsigned)(uintptr_t)&sB1[cur ^ 1][0], Wtb + (size_t)p0 * HD + kn, 32, 128, HD);
        tdm_load_2d((unsigned)(uintptr_t)&sB2[cur ^ 1][0], Wjb + (size_t)p0 * HD + kn, 32, 128, HD);
        tdm_load_2d((unsigned)(uintptr_t)&sB3[cur ^ 1][0], Brb + (size_t)p0 * HD + kn, 32, 128, HD);
        tdm_load_2d((unsigned)(uintptr_t)&sB4[cur ^ 1][0], Bib + (size_t)p0 * HD + kn, 32, 128, HD);
        __builtin_amdgcn_s_wait_tensorcnt(7);
      } else {
        __builtin_amdgcn_s_wait_tensorcnt(0);
      }
    }
    __syncthreads();
#else
    __syncthreads();
    { int m = t >> 2, c = (t & 3) * 8;
      *(uint4*)(&sA1[cur][0] + m * 32 + c) =
          *(const uint4*)(feats + (size_t)(m0 + m) * HD + k0 + c);
      *(uint4*)(&sA2[cur][0] + m * 32 + c) =
          *(const uint4*)(Xb + (size_t)(m0 + m) * HD + k0 + c); }
    { int nn = t >> 1, c = (t & 1) * 16;
      size_t off = (size_t)(p0 + nn) * HD + k0 + c;
      int d = nn * 32 + c;
      *(uint4*)(&sB0[cur][0]+d) = *(const uint4*)(Wrb+off);
      *(uint4*)(&sB0[cur][0]+d+8) = *(const uint4*)(Wrb+off+8);
      *(uint4*)(&sB1[cur][0]+d) = *(const uint4*)(Wtb+off);
      *(uint4*)(&sB1[cur][0]+d+8) = *(const uint4*)(Wtb+off+8);
      *(uint4*)(&sB2[cur][0]+d) = *(const uint4*)(Wjb+off);
      *(uint4*)(&sB2[cur][0]+d+8) = *(const uint4*)(Wjb+off+8);
      *(uint4*)(&sB3[cur][0]+d) = *(const uint4*)(Brb+off);
      *(uint4*)(&sB3[cur][0]+d+8) = *(const uint4*)(Brb+off+8);
      *(uint4*)(&sB4[cur][0]+d) = *(const uint4*)(Bib+off);
      *(uint4*)(&sB4[cur][0]+d+8) = *(const uint4*)(Bib+off+8); }
    __syncthreads();
#endif
    ABf a1[2], a2[2];
    #pragma unroll
    for (int ti = 0; ti < 2; ++ti) {
      const unsigned short* p1 = &sA1[cur][0] + (wm * 32 + ti * 16 + ln) * 32;
      const unsigned short* p2 = &sA2[cur][0] + (wm * 32 + ti * 16 + ln) * 32;
      a1[ti].q[0] = *(const uint4*)(p1 + hi * 8);
      a1[ti].q[1] = *(const uint4*)(p1 + 16 + hi * 8);
      a2[ti].q[0] = *(const uint4*)(p2 + hi * 8);
      a2[ti].q[1] = *(const uint4*)(p2 + 16 + hi * 8);
    }
    #pragma unroll
    for (int tj = 0; tj < 2; ++tj) {
      const int nb = (wn * 32 + tj * 16 + ln) * 32 + hi * 8;
      ABf b;
      b.q[0] = *(const uint4*)(&sB0[cur][0] + nb); b.q[1] = *(const uint4*)(&sB0[cur][0] + nb + 16);
      #pragma unroll
      for (int ti = 0; ti < 2; ++ti) aR[ti][tj] = wmma_bf16(a1[ti], b, aR[ti][tj]);
      b.q[0] = *(const uint4*)(&sB1[cur][0] + nb); b.q[1] = *(const uint4*)(&sB1[cur][0] + nb + 16);
      #pragma unroll
      for (int ti = 0; ti < 2; ++ti) aT[ti][tj] = wmma_bf16(a1[ti], b, aT[ti][tj]);
      b.q[0] = *(const uint4*)(&sB2[cur][0] + nb); b.q[1] = *(const uint4*)(&sB2[cur][0] + nb + 16);
      #pragma unroll
      for (int ti = 0; ti < 2; ++ti) aI[ti][tj] = wmma_bf16(a1[ti], b, aI[ti][tj]);
      b.q[0] = *(const uint4*)(&sB3[cur][0] + nb); b.q[1] = *(const uint4*)(&sB3[cur][0] + nb + 16);
      #pragma unroll
      for (int ti = 0; ti < 2; ++ti) aBr[ti][tj] = wmma_bf16(a2[ti], b, aBr[ti][tj]);
      b.q[0] = *(const uint4*)(&sB4[cur][0] + nb); b.q[1] = *(const uint4*)(&sB4[cur][0] + nb + 16);
      #pragma unroll
      for (int ti = 0; ti < 2; ++ti) aBi[ti][tj] = wmma_bf16(a2[ti], b, aBi[ti][tj]);
    }
    __syncthreads();
  }

  // epilogue: spectral params -> IMEX1 scan coefficients
  #pragma unroll
  for (int tj = 0; tj < 2; ++tj) {
    int p = p0 + wn * 32 + tj * 16 + ln;
    float rb = r_base[p], tb = th_base[p], ib = inj_logit[p];
    float dt = 0.02f + 0.98f * sigmoidf(dt_base[p]);
    float dtc = fmaxf(dt, 1e-6f);
    #pragma unroll
    for (int ti = 0; ti < 2; ++ti)
      #pragma unroll
      for (int v = 0; v < 8; ++v) {
        int m = m0 + wm * 32 + ti * 16 + v + hi * 8;
        size_t o = (size_t)m * PD + p;
        float r  = sigmoidf(rb + aR[ti][tj][v]);
        float th = PI_F * tanhf(tb + aT[ti][tj][v]);
        float r2 = fmaxf(r * r, 1e-8f);
        float ct = __cosf(th);
        float A  = (r2 - 2.0f * r * ct + 1.0f) / (dtc * dtc * r2);
        float G  = (1.0f - r2) / (dtc * r2);
        float S  = 1.0f / (1.0f + dtc * G);
        float gate = sigmoidf(ib + aI[ti][tj][v]);
        cS[o]  = S;
        cM[o]  = S * dtc * A;
        cSD[o] = S * dtc;
        bRe[o] = aBr[ti][tj][v] * gate;
        bIm[o] = aBi[ti][tj][v] * gate;
      }
  }
}

// ============================================================
// Kernel 4: D-LinOSS IMEX1 scan.  One thread per (b,p), coalesced
// along p:  z = S*z - M*x + SD*bu ;  x += dt*z
// ============================================================
__global__ __launch_bounds__(256) void k_scan(
    const float* __restrict__ cS, const float* __restrict__ cM,
    const float* __restrict__ cSD, const float* __restrict__ bRe,
    const float* __restrict__ bIm, const float* __restrict__ dt_base,
    unsigned short* __restrict__ xb)
{
  int idx = blockIdx.x * 256 + threadIdx.x;
  if (idx >= BSZ * PD) return;
  int b = idx / PD, p = idx - b * PD;
  float dt = 0.02f + 0.98f * sigmoidf(dt_base[p]);
  dt = fmaxf(dt, 1e-6f);
  float zr = 0.f, zi = 0.f, xr = 0.f, xi = 0.f;
  size_t base  = (size_t)b * LENT * PD + p;
  size_t obase = (size_t)b * LENT * (2 * PD) + p;
  for (int l = 0; l < LENT; ++l) {
    size_t k = base + (size_t)l * PD;
    float S = cS[k], M = cM[k], SD = cSD[k];
    float br = bRe[k], bi = bIm[k];
    zr = S * zr - M * xr + SD * br;
    zi = S * zi - M * xi + SD * bi;
    xr += dt * zr;
    xi += dt * zi;
    size_t o = obase + (size_t)l * (2 * PD);
    xb[o]      = f2bf(xr);
    xb[o + PD] = f2bf(xi);
  }
}

// ============================================================
// Kernel 5: out = xb @ [Cre|-Cim]^T + X*D   (f32 out)
// ============================================================
__global__ __launch_bounds__(256) void k_out_gemm(
    const unsigned short* __restrict__ xb, const unsigned short* __restrict__ Ccb,
    const float* __restrict__ X, const float* __restrict__ Dv,
    float* __restrict__ out)
{
  __shared__ __align__(16) unsigned short sA[2][64 * 32];
  __shared__ __align__(16) unsigned short sB[2][128 * 32];
  const int m0 = blockIdx.x * 64;
  const int n0 = blockIdx.y * 128;
  const int t  = threadIdx.x;
  const int lane = t & 31, wid = t >> 5;
  const int wm = wid >> 2, wn = wid & 3;
  const int hi = lane >> 4, ln = lane & 15;

  v8f acc[2][2];
  #pragma unroll
  for (int i = 0; i < 2; ++i)
    #pragma unroll
    for (int j = 0; j < 2; ++j) acc[i][j] = (v8f){};

#if USE_TDM
  if (wid == 0) {
    tdm_load_2d((unsigned)(uintptr_t)&sA[0][0], xb + (size_t)m0 * (2 * PD), 32, 64, 2 * PD);
    tdm_load_2d((unsigned)(uintptr_t)&sB[0][0], Ccb + (size_t)n0 * (2 * PD), 32, 128, 2 * PD);
  }
#endif
  for (int kc = 0; kc < (2 * PD) / 32; ++kc) {
    const int cur = kc & 1;
    const int k0 = kc * 32;
#if USE_TDM
    if (wid == 0) {
      if (kc + 1 < (2 * PD) / 32) {
        tdm_load_2d((unsigned)(uintptr_t)&sA[cur ^ 1][0],
                    xb + (size_t)m0 * (2 * PD) + k0 + 32, 32, 64, 2 * PD);
        tdm_load_2d((unsigned)(uintptr_t)&sB[cur ^ 1][0],
                    Ccb + (size_t)n0 * (2 * PD) + k0 + 32, 32, 128, 2 * PD);
        __builtin_amdgcn_s_wait_tensorcnt(2);
      } else {
        __builtin_amdgcn_s_wait_tensorcnt(0);
      }
    }
    __syncthreads();
#else
    __syncthreads();
    { int m = t >> 2, c = (t & 3) * 8;
      *(uint4*)(&sA[cur][0] + m * 32 + c) =
          *(const uint4*)(xb + (size_t)(m0 + m) * (2 * PD) + k0 + c); }
    { int nn = t >> 1, c = (t & 1) * 16;
      const unsigned short* s = Ccb + (size_t)(n0 + nn) * (2 * PD) + k0 + c;
      unsigned short* d = &sB[cur][0] + nn * 32 + c;
      *(uint4*)(d) = *(const uint4*)(s);
      *(uint4*)(d + 8) = *(const uint4*)(s + 8); }
    __syncthreads();
#endif
    const unsigned short* cA = &sA[cur][0];
    const unsigned short* cB = &sB[cur][0];
    ABf a[2];
    #pragma unroll
    for (int ti = 0; ti < 2; ++ti) {
      const unsigned short* pa = cA + (wm * 32 + ti * 16 + ln) * 32;
      a[ti].q[0] = *(const uint4*)(pa + hi * 8);
      a[ti].q[1] = *(const uint4*)(pa + 16 + hi * 8);
    }
    #pragma unroll
    for (int tj = 0; tj < 2; ++tj) {
      ABf b;
      const unsigned short* pb = cB + (wn * 32 + tj * 16 + ln) * 32 + hi * 8;
      b.q[0] = *(const uint4*)(pb);
      b.q[1] = *(const uint4*)(pb + 16);
      #pragma unroll
      for (int ti = 0; ti < 2; ++ti)
        acc[ti][tj] = wmma_bf16(a[ti], b, acc[ti][tj]);
    }
    __syncthreads();
  }

  #pragma unroll
  for (int tj = 0; tj < 2; ++tj) {
    int n = n0 + wn * 32 + tj * 16 + ln;
    float dv = Dv[n];
    #pragma unroll
    for (int ti = 0; ti < 2; ++ti)
      #pragma unroll
      for (int v = 0; v < 8; ++v) {
        int m = m0 + wm * 32 + ti * 16 + v + hi * 8;
        out[(size_t)m * HD + n] = acc[ti][tj][v] + X[(size_t)m * HD + n] * dv;
      }
  }
}

// ============================================================
extern "C" void kernel_launch(void* const* d_in, const int* in_sizes, int n_in,
                              void* d_out, int out_size, void* d_ws, size_t ws_size,
                              hipStream_t stream)
{
  (void)in_sizes; (void)n_in; (void)out_size; (void)ws_size;
  const float* X   = (const float*)d_in[0];   // (16,2048,768)
  const float* Wl  = (const float*)d_in[1];   // (768,768)
  const float* bl  = (const float*)d_in[2];   // (768,)
  const float* cw  = (const float*)d_in[3];   // (768,1,4)
  const float* cb  = (const float*)d_in[4];   // (768,)
  const float* Wr  = (const float*)d_in[5];   // (384,768)
  const float* Wth = (const float*)d_in[6];   // (384,768)
  const float* Wnj = (const float*)d_in[7];   // (384,768)
  const float* rb  = (const float*)d_in[8];   // (384,)
  const float* tb  = (const float*)d_in[9];   // (384,)
  const float* db  = (const float*)d_in[10];  // (384,)
  const float* il  = (const float*)d_in[11];  // (384,)
  const float* Bp  = (const float*)d_in[12];  // (384,768,2)
  const float* Cp  = (const float*)d_in[13];  // (768,384,2)
  const float* Dv  = (const float*)d_in[14];  // (768,)
  float* out = (float*)d_out;

  char* ws = (char*)d_ws;
  constexpr size_t SEG = (size_t)NTOK * HD * 2;   // 48 MiB; == NTOK*PD*4 too
  unsigned short* Xbf    = (unsigned short*)(ws + 0 * SEG);  // bf16 N x H
  unsigned short* feats0 = (unsigned short*)(ws + 1 * SEG);  // bf16 N x H
  unsigned short* feats  = (unsigned short*)(ws + 2 * SEG);  // bf16 N x H
  float* cS  = (float*)(ws + 3 * SEG);                       // f32  N x P
  float* cM  = (float*)(ws + 4 * SEG);
  float* cSD = (float*)(ws + 5 * SEG);
  float* bRe = (float*)(ws + 6 * SEG);
  float* bIm = (float*)(ws + 7 * SEG);
  unsigned short* xbuf = (unsigned short*)(ws + 8 * SEG);    // bf16 N x 2P
  unsigned short* wb   = (unsigned short*)(ws + 9 * SEG);    // packed bf16 weights
  unsigned short* Wlb = wb;
  unsigned short* Wrb = wb + O_WR;
  unsigned short* Wtb = wb + O_WT;
  unsigned short* Wjb = wb + O_WJ;
  unsigned short* Brb = wb + O_BR;
  unsigned short* Bib = wb + O_BI;
  unsigned short* Ccb = wb + O_CC;

  k_cvt_x<<<dim3((NTOK * HD) / 8 / 256), 256, 0, stream>>>(X, Xbf);
  k_prep_w<<<dim3((W_TOTAL + 255) / 256), 256, 0, stream>>>(Wl, Wr, Wth, Wnj, Bp, Cp, wb);
  k_enc_gemm<<<dim3(NTOK / 64, HD / 128), 256, 0, stream>>>(Xbf, Wlb, bl, feats0);
  k_conv<<<dim3((NTOK * (HD / 8)) / 256), 256, 0, stream>>>(feats0, cw, cb, feats);
  k_sel_gemm<<<dim3(NTOK / 64, PD / 128), 256, 0, stream>>>(
      feats, Xbf, Wrb, Wtb, Wjb, Brb, Bib, rb, tb, db, il, cS, cM, cSD, bRe, bIm);
  k_scan<<<dim3((BSZ * PD + 255) / 256), 256, 0, stream>>>(cS, cM, cSD, bRe, bIm, db, xbuf);
  k_out_gemm<<<dim3(NTOK / 64, HD / 128), 256, 0, stream>>>(xbuf, Ccb, X, Dv, out);
}
